// NonLocalAttention_91070486544492
// MI455X (gfx1250) — matrix operand order repaired and checked by reading, not compiled, not run on hardware
//
#include <hip/hip_runtime.h>

// ---------------------------------------------------------------------------
// NonLocalAttention pyramid for MI455X (gfx1250, wave32, WMMA).
// Stage-1 q3^T k3 (M=N=2304, K=9 padded to 32) runs on v_wmma_f32_16x16x32_f16;
// the 2304x2304 score matrix never touches HBM. Top-3 tracking uses packed
// (score,index) u64 keys and compare-exchange networks -> pure v_cmp_lt_u64 +
// v_cndmask (no EXEC-mask branching), distributed over all 32 lanes.
// ---------------------------------------------------------------------------

typedef __attribute__((ext_vector_type(16))) _Float16 v16h;
typedef __attribute__((ext_vector_type(8)))  float    v8f;

static constexpr int B_  = 8;
static constexpr int H1_ = 192, W1_ = 192, C1_ = 3,  N1_ = H1_ * W1_;   // 36864
static constexpr int H2_ = 96,  W2_ = 96,  C2_ = 6,  N2_ = H2_ * W2_;   // 9216
static constexpr int H3_ = 48,  W3_ = 48,  C3_ = 9,  N3_ = H3_ * W3_;   // 2304
static constexpr int NT3_ = N3_ / 16;                                   // 144 tiles

// ---- packed (score,index) keys: larger key == larger score, tie -> lower n --
__device__ __forceinline__ unsigned long long nla_pack(float s, unsigned n) {
  unsigned u = __float_as_uint(s);
  u ^= 0x80000000u | (unsigned)((int)u >> 31);   // monotonic float -> uint
  return ((unsigned long long)u << 32) | (unsigned)~n;
}
__device__ __forceinline__ unsigned nla_idx(unsigned long long k) {
  return ~(unsigned)k;
}
// Insert x into sorted t0 >= t1 >= t2 (3 compare-exchanges, branchless).
__device__ __forceinline__ void nla_top3(unsigned long long& t0,
                                         unsigned long long& t1,
                                         unsigned long long& t2,
                                         unsigned long long x) {
  t2 = (t2 > x) ? t2 : x;
  unsigned long long m1 = (t1 > t2) ? t1 : t2;
  unsigned long long n1 = (t1 > t2) ? t2 : t1;
  t1 = m1; t2 = n1;
  unsigned long long m0 = (t0 > t1) ? t0 : t1;
  unsigned long long n0 = (t0 > t1) ? t1 : t0;
  t0 = m0; t1 = n0;
}

// ---------------- kernel 1: q1/k1 = 3x3 channel projections of x -----------
__global__ void nla_qk1(const float* __restrict__ x,
                        const float* __restrict__ wq1, const float* __restrict__ wk1,
                        float* __restrict__ q1, float* __restrict__ k1) {
  int idx = blockIdx.x * blockDim.x + threadIdx.x;
  if (idx >= B_ * N1_) return;
  int b = idx / N1_, p = idx % N1_;
  const float* xb = x + ((size_t)b * C1_) * N1_ + p;
  float c0 = xb[0], c1 = xb[N1_], c2 = xb[2 * N1_];
#pragma unroll
  for (int o = 0; o < 3; ++o) {
    q1[((size_t)b * 3 + o) * N1_ + p] = wq1[o*3+0]*c0 + wq1[o*3+1]*c1 + wq1[o*3+2]*c2;
    k1[((size_t)b * 3 + o) * N1_ + p] = wk1[o*3+0]*c0 + wk1[o*3+1]*c1 + wk1[o*3+2]*c2;
  }
}

// ---------------- kernel 2: conv1 (3->6, s2 p1) fused with q2/k2 -----------
__global__ void nla_conv1(const float* __restrict__ x, const float* __restrict__ w1,
                          const float* __restrict__ wq2, const float* __restrict__ wk2,
                          float* __restrict__ x2, float* __restrict__ q2,
                          float* __restrict__ k2) {
  int idx = blockIdx.x * blockDim.x + threadIdx.x;
  if (idx >= B_ * N2_) return;
  int b = idx / N2_, p = idx % N2_;
  int y = p / W2_, xo = p % W2_;
  float acc[6] = {0.f, 0.f, 0.f, 0.f, 0.f, 0.f};
  for (int c = 0; c < 3; ++c)
    for (int ky = 0; ky < 3; ++ky) {
      int iy = 2 * y - 1 + ky;
      if (iy < 0 || iy >= H1_) continue;
      for (int kx = 0; kx < 3; ++kx) {
        int ix = 2 * xo - 1 + kx;
        if (ix < 0 || ix >= W1_) continue;
        float v = x[((size_t)b * 3 + c) * N1_ + iy * W1_ + ix];
#pragma unroll
        for (int o = 0; o < 6; ++o) acc[o] += v * w1[((o * 3 + c) * 3 + ky) * 3 + kx];
      }
    }
#pragma unroll
  for (int o = 0; o < 6; ++o) x2[((size_t)b * 6 + o) * N2_ + p] = acc[o];
#pragma unroll
  for (int o = 0; o < 6; ++o) {
    float q = 0.f, k = 0.f;
#pragma unroll
    for (int c = 0; c < 6; ++c) { q += wq2[o*6+c]*acc[c]; k += wk2[o*6+c]*acc[c]; }
    q2[((size_t)b * 6 + o) * N2_ + p] = q;
    k2[((size_t)b * 6 + o) * N2_ + p] = k;
  }
}

// ------- kernel 3: conv2 (6->9, s2 p1) fused with q3/k3, WMMA layouts ------
// q3A: A-matrix (16x32 f16) layout: [b][mtile][lane 0..31][16 halves]
//   lane<16  (row m=lane):    e0..7 = ch0..7,  e8..15 = ch16..23 (=0)
//   lane>=16 (row m=lane-16): e0..7 = ch8..15 (ch8 real), e8..15 = ch24..31 (=0)
// k3B: B-matrix (32x16 f16) layout: [b][ntile][lane][16 halves]
//   lane<16  (col n=lane):    e0..7 = ch0..7,  e8..15 = ch16..23 (=0)
//   lane>=16 (col n=lane-16): e0..7 = ch8..15 (ch8 real), e8..15 = ch24..31 (=0)
__global__ void nla_conv2(const float* __restrict__ x2, const float* __restrict__ w2,
                          const float* __restrict__ wq3, const float* __restrict__ wk3,
                          _Float16* __restrict__ q3A, _Float16* __restrict__ k3B) {
  int idx = blockIdx.x * blockDim.x + threadIdx.x;
  if (idx >= B_ * N3_) return;
  int b = idx / N3_, p = idx % N3_;
  int y = p / W3_, xo = p % W3_;
  float acc[9] = {0.f,0.f,0.f,0.f,0.f,0.f,0.f,0.f,0.f};
  for (int c = 0; c < 6; ++c)
    for (int ky = 0; ky < 3; ++ky) {
      int iy = 2 * y - 1 + ky;
      if (iy < 0 || iy >= H2_) continue;
      for (int kx = 0; kx < 3; ++kx) {
        int ix = 2 * xo - 1 + kx;
        if (ix < 0 || ix >= W2_) continue;
        float v = x2[((size_t)b * 6 + c) * N2_ + iy * W2_ + ix];
#pragma unroll
        for (int o = 0; o < 9; ++o) acc[o] += v * w2[((o * 6 + c) * 3 + ky) * 3 + kx];
      }
    }
  float q[9], k[9];
#pragma unroll
  for (int o = 0; o < 9; ++o) {
    float qa = 0.f, ka = 0.f;
#pragma unroll
    for (int c = 0; c < 9; ++c) { qa += wq3[o*9+c]*acc[c]; ka += wk3[o*9+c]*acc[c]; }
    q[o] = qa; k[o] = ka;
  }
  int tile = p >> 4, lp = p & 15;
  _Float16* qa = q3A + (((size_t)b * NT3_ + tile) * 32) * 16;
  _Float16* kb = k3B + (((size_t)b * NT3_ + tile) * 32) * 16;
#pragma unroll
  for (int e = 0; e < 16; ++e) {
    qa[lp * 16 + e]        = (e < 8) ? (_Float16)q[e] : (_Float16)0.f;
    qa[(16 + lp) * 16 + e] = (e == 0) ? (_Float16)q[8] : (_Float16)0.f;
    kb[lp * 16 + e]        = (e < 8) ? (_Float16)k[e] : (_Float16)0.f;
    kb[(16 + lp) * 16 + e] = (e == 0) ? (_Float16)k[8] : (_Float16)0.f;
  }
}

// ---------------- kernel 4: stage-1 WMMA scores + branchless top-3 ---------
__global__ __launch_bounds__(32) void nla_stage1_wmma(const _Float16* __restrict__ q3A,
                                                      const _Float16* __restrict__ k3B,
                                                      int* __restrict__ idx3) {
  const int b = blockIdx.y, mt = blockIdx.x, lane = threadIdx.x;
  const int col = lane & 15;
  const int rbase = (lane >> 4) * 8;

  v16h a = *(const v16h*)(q3A + (((size_t)b * NT3_ + mt) * 32 + lane) * 16);

  // 8 private top-3 trackers per lane: row (rbase+r), column slice `col`.
  unsigned long long t0[8], t1[8], t2[8];
#pragma unroll
  for (int r = 0; r < 8; ++r) { t0[r] = 0ull; t1[r] = 0ull; t2[r] = 0ull; }

  for (int nt = 0; nt < NT3_; ++nt) {
    const _Float16* kp = k3B + (((size_t)b * NT3_ + nt) * 32 + lane) * 16;
    __builtin_prefetch(kp + 32 * 16, 0, 1);  // speculative global_prefetch_b8
    v16h bm = *(const v16h*)kp;
    v8f c = {};
    c = __builtin_amdgcn_wmma_f32_16x16x32_f16(false, a, false, bm,
                                               (short)0, c, false, false);
#pragma unroll
    for (int r = 0; r < 8; ++r)
      nla_top3(t0[r], t1[r], t2[r], nla_pack(c[r], (unsigned)nt));
  }

  // Dump trackers with global column index folded into the key's low bits,
  // then merge the 16 column-slices per row (ties resolved by the key itself).
  __shared__ unsigned long long skey[16][16][3];
#pragma unroll
  for (int r = 0; r < 8; ++r) {
    int row = rbase + r;
    unsigned long long hi;
    hi = t0[r] >> 32; skey[row][col][0] = (hi << 32) | (unsigned)~(nla_idx(t0[r]) * 16 + col);
    hi = t1[r] >> 32; skey[row][col][1] = (hi << 32) | (unsigned)~(nla_idx(t1[r]) * 16 + col);
    hi = t2[r] >> 32; skey[row][col][2] = (hi << 32) | (unsigned)~(nla_idx(t2[r]) * 16 + col);
  }
  __syncthreads();

  if (lane < 16) {
    unsigned long long b0 = 0ull, b1 = 0ull, b2 = 0ull;
    for (int cc = 0; cc < 16; ++cc) {
#pragma unroll
      for (int k = 0; k < 3; ++k)
        nla_top3(b0, b1, b2, skey[lane][cc][k]);
    }
    int m = mt * 16 + lane;
    int* o = idx3 + ((size_t)b * N3_ + m) * 3;
    o[0] = (int)nla_idx(b0); o[1] = (int)nla_idx(b1); o[2] = (int)nla_idx(b2);
  }
}

// expand_indices column pattern (scale=2): from top-3 bases v0,v1,v2 build the
// 12-vector [v0+2W, v0+4W, v1+4W, v1+4W, v2+4W, v2+4W,
//            v0+2W, v0+2W, v1+2W, v1+2W, v2+2W, v2+2W], then sort ascending.
__device__ __forceinline__ void nla_expand12(const int v[3], int W, int e[12]) {
  const int tw = 2 * W, fw = 4 * W;
  e[0] = v[0] + tw; e[1] = v[0] + fw; e[2]  = v[1] + fw; e[3]  = v[1] + fw;
  e[4] = v[2] + fw; e[5] = v[2] + fw; e[6]  = v[0] + tw; e[7]  = v[0] + tw;
  e[8] = v[1] + tw; e[9] = v[1] + tw; e[10] = v[2] + tw; e[11] = v[2] + tw;
#pragma unroll
  for (int pass = 0; pass < 12; ++pass) {
#pragma unroll
    for (int q = (pass & 1); q < 11; q += 2) {
      int lo = (e[q] < e[q + 1]) ? e[q] : e[q + 1];
      int hi = (e[q] < e[q + 1]) ? e[q + 1] : e[q];
      e[q] = lo; e[q + 1] = hi;
    }
  }
}

// ---------------- kernel 5: stage-2 gather attention -> local top-3 --------
__global__ void nla_stage2(const float* __restrict__ q2, const float* __restrict__ k2,
                           const int* __restrict__ idx3, int* __restrict__ idx2) {
  int idx = blockIdx.x * blockDim.x + threadIdx.x;
  if (idx >= B_ * N2_) return;
  int b = idx / N2_, r = idx % N2_;
  int nsrc = (r % (2 * N3_)) >> 1;
  const int* t = idx3 + ((size_t)b * N3_ + nsrc) * 3;
  int add1 = (nsrc == 0) ? 1 : 0;
  int v[3];
#pragma unroll
  for (int kk = 0; kk < 3; ++kk) {
    int tt = t[kk], i = tt % W3_;
    v[kk] = (tt - i) * 4 + i * 2 + add1;  // j*scale^2 + i*scale (+row-0 quirk)
  }
  int e[12];
  nla_expand12(v, W3_, e);

  float qv[6];
#pragma unroll
  for (int c = 0; c < 6; ++c) qv[c] = q2[((size_t)b * 6 + c) * N2_ + r];

  unsigned long long b0 = 0ull, b1 = 0ull, b2 = 0ull;
#pragma unroll
  for (int kk = 0; kk < 12; ++kk) {
    int g = e[kk]; g = (g < 0) ? 0 : (g >= N2_ ? N2_ - 1 : g);  // JAX clips OOB
    float s = 0.f;
#pragma unroll
    for (int c = 0; c < 6; ++c) s += qv[c] * k2[((size_t)b * 6 + c) * N2_ + g];
    nla_top3(b0, b1, b2, nla_pack(s, (unsigned)kk));
  }
  int* o = idx2 + ((size_t)b * N2_ + r) * 3;
  o[0] = (int)nla_idx(b0); o[1] = (int)nla_idx(b1); o[2] = (int)nla_idx(b2);
}

// ---------------- kernel 6: stage-3 + output (replicates local-index bug) --
__global__ void nla_stage3(const float* __restrict__ q1, const float* __restrict__ k1,
                           const int* __restrict__ idx2, float* __restrict__ out) {
  int idx = blockIdx.x * blockDim.x + threadIdx.x;
  if (idx >= B_ * N1_) return;
  int b = idx / N1_, r = idx % N1_;
  int nsrc = (r % (2 * N2_)) >> 1;
  const int* t = idx2 + ((size_t)b * N2_ + nsrc) * 3;
  int add1 = (nsrc == 0) ? 1 : 0;
  int v[3];
#pragma unroll
  for (int kk = 0; kk < 3; ++kk) v[kk] = 2 * t[kk] + add1;  // t<12<W2 => j=0, i=t
  int e[12];
  nla_expand12(v, W2_, e);

  float qv[3];
#pragma unroll
  for (int c = 0; c < 3; ++c) qv[c] = q1[((size_t)b * 3 + c) * N1_ + r];

  unsigned long long b0 = 0ull, b1 = 0ull, b2 = 0ull;
#pragma unroll
  for (int kk = 0; kk < 12; ++kk) {
    int g = e[kk]; g = (g < 0) ? 0 : (g >= N1_ ? N1_ - 1 : g);
    float s = 0.f;
#pragma unroll
    for (int c = 0; c < 3; ++c) s += qv[c] * k1[((size_t)b * 3 + c) * N1_ + g];
    nla_top3(b0, b1, b2, nla_pack(s, (unsigned)kk));
  }
  // Output stage: locals l0..l2 used as ABSOLUTE k1 columns (faithful bug).
  const float scale = 0.5773502691896258f;  // 1/sqrt(3)
  int ll[3] = { (int)nla_idx(b0), (int)nla_idx(b1), (int)nla_idx(b2) };
  float fs[3];
#pragma unroll
  for (int j = 0; j < 3; ++j) {
    float s = 0.f;
#pragma unroll
    for (int c = 0; c < 3; ++c) s += qv[c] * k1[((size_t)b * 3 + c) * N1_ + ll[j]];
    fs[j] = s * scale;
  }
  float mx = fmaxf(fs[0], fmaxf(fs[1], fs[2]));
  float w0 = __expf(fs[0] - mx), w1 = __expf(fs[1] - mx), w2 = __expf(fs[2] - mx);
  float inv = 1.f / (w0 + w1 + w2);
#pragma unroll
  for (int c = 0; c < 3; ++c) {
    const float* kc = k1 + ((size_t)b * 3 + c) * N1_;
    out[((size_t)b * 3 + c) * N1_ + r] = (w0 * kc[ll[0]] + w1 * kc[ll[1]] + w2 * kc[ll[2]]) * inv;
  }
}

// ---------------------------------------------------------------------------
extern "C" void kernel_launch(void* const* d_in, const int* in_sizes, int n_in,
                              void* d_out, int out_size, void* d_ws, size_t ws_size,
                              hipStream_t stream) {
  const float* x   = (const float*)d_in[0];
  const float* w1  = (const float*)d_in[1];
  const float* w2  = (const float*)d_in[2];
  const float* wq3 = (const float*)d_in[3];
  const float* wk3 = (const float*)d_in[4];
  const float* wq2 = (const float*)d_in[5];
  const float* wk2 = (const float*)d_in[6];
  const float* wq1 = (const float*)d_in[7];
  const float* wk1 = (const float*)d_in[8];
  float* out = (float*)d_out;

  char* ws = (char*)d_ws;
  size_t off = 0;
  auto take = [&](size_t bytes) -> char* {
    char* p = ws + off;
    off += (bytes + 255) & ~(size_t)255;
    return p;
  };
  float*    q1   = (float*)take((size_t)B_ * 3 * N1_ * sizeof(float));
  float*    k1   = (float*)take((size_t)B_ * 3 * N1_ * sizeof(float));
  float*    x2   = (float*)take((size_t)B_ * 6 * N2_ * sizeof(float));
  float*    q2   = (float*)take((size_t)B_ * 6 * N2_ * sizeof(float));
  float*    k2   = (float*)take((size_t)B_ * 6 * N2_ * sizeof(float));
  _Float16* q3A  = (_Float16*)take((size_t)B_ * NT3_ * 32 * 16 * sizeof(_Float16));
  _Float16* k3B  = (_Float16*)take((size_t)B_ * NT3_ * 32 * 16 * sizeof(_Float16));
  int*      idx3 = (int*)take((size_t)B_ * N3_ * 3 * sizeof(int));
  int*      idx2 = (int*)take((size_t)B_ * N2_ * 3 * sizeof(int));
  (void)ws_size; (void)in_sizes; (void)n_in; (void)out_size;

  const int TB = 256;
  nla_qk1   <<<(B_ * N1_ + TB - 1) / TB, TB, 0, stream>>>(x, wq1, wk1, q1, k1);
  nla_conv1 <<<(B_ * N2_ + TB - 1) / TB, TB, 0, stream>>>(x, w1, wq2, wk2, x2, q2, k2);
  nla_conv2 <<<(B_ * N3_ + TB - 1) / TB, TB, 0, stream>>>(x2, w2, wq3, wk3, q3A, k3B);
  nla_stage1_wmma<<<dim3(NT3_, B_), 32, 0, stream>>>(q3A, k3B, idx3);
  nla_stage2<<<(B_ * N2_ + TB - 1) / TB, TB, 0, stream>>>(q2, k2, idx3, idx2);
  nla_stage3<<<(B_ * N1_ + TB - 1) / TB, TB, 0, stream>>>(q1, k1, idx2, out);
}
